// TransformerConvLayer_65609920413936
// MI455X (gfx1250) — compile-verified, matrix operand order checked by reading, not run on hardware
//
#include <hip/hip_runtime.h>
#include <math.h>

// Problem constants (match reference)
#define BDIM 2
#define NDIM 2048
#define DDIM 128

typedef __attribute__((ext_vector_type(16))) _Float16 v16h;
typedef __attribute__((ext_vector_type(8)))  _Float16 v8h;
typedef __attribute__((ext_vector_type(8)))  float    v8f;

__device__ __forceinline__ v8f zero_v8f() {
  v8f z;
#pragma unroll
  for (int t = 0; t < 8; ++t) z[t] = 0.0f;
  return z;
}

__device__ __forceinline__ v8f wmma16(v16h a, v16h b, v8f c) {
  // D = A(16x32 f16) * B(32x16 f16) + C(16x16 f32)
  return __builtin_amdgcn_wmma_f32_16x16x32_f16(false, a, false, b, (short)0, c,
                                                false, false);
}

// Fragment load from a pre-offset pointer (caller already added the per-lane
// offset (lane&15)*ld + (lane>>4)*8). Per CDNA5 ISA f16 A/B layout:
// halves 0..7 = K of (lane>>4)*8..+7 ; halves 8..15 = K of 16+(lane>>4)*8..+7.
__device__ __forceinline__ v16h ldfrag_pre(const _Float16* p) {
  v8h lo = *(const v8h*)p;
  v8h hi = *(const v8h*)(p + 16);
  v16h r;
#pragma unroll
  for (int t = 0; t < 8; ++t) { r[t] = lo[t]; r[t + 8] = hi[t]; }
  return r;
}

// Same, but fp32 source converted on the fly.
__device__ __forceinline__ v16h ldfrag_pre_f32(const float* p) {
  v16h r;
#pragma unroll
  for (int t = 0; t < 8; ++t) {
    r[t]     = (_Float16)p[t];
    r[t + 8] = (_Float16)p[t + 16];
  }
  return r;
}

// ---------------------------------------------------------------------------
// e_t[b, i(dst), j(src)] = edge[b, j, i] * A[b, j, i]
// ---------------------------------------------------------------------------
__global__ void prep_et_kernel(const float* __restrict__ edge,
                               const float* __restrict__ A,
                               float* __restrict__ e_t) {
  size_t idx = (size_t)blockIdx.x * blockDim.x + threadIdx.x;
  size_t tot = (size_t)BDIM * NDIM * NDIM;
  if (idx >= tot) return;
  int b = (int)(idx / ((size_t)NDIM * NDIM));
  int r = (int)(idx % ((size_t)NDIM * NDIM));
  int i = r / NDIM;  // dst
  int j = r % NDIM;  // src
  size_t src = ((size_t)b * NDIM + j) * NDIM + i;
  e_t[idx] = edge[src] * A[src];
}

// ---------------------------------------------------------------------------
// Weight prep: Wt[c, k] = (f16) W[k, c]   (W is fin=128 x C row-major fp32)
// Makes the GEMM B-operand a contiguous 16B fragment load.
// ---------------------------------------------------------------------------
__global__ void prep_wt_kernel(const float* __restrict__ W,
                               _Float16* __restrict__ Wt, int C) {
  int idx = blockIdx.x * blockDim.x + threadIdx.x;
  if (idx >= C * DDIM) return;
  int c = idx / DDIM;
  int k = idx % DDIM;
  Wt[idx] = (_Float16)W[(size_t)k * C + c];
}

// ---------------------------------------------------------------------------
// Layer 0 projections: fin = 1 -> rank-1 outer products.
// ---------------------------------------------------------------------------
__global__ void layer0_kernel(const float* __restrict__ node,
                              const float* __restrict__ Wq,
                              const float* __restrict__ Wk,
                              const float* __restrict__ Wv,
                              const float* __restrict__ Ws,
                              _Float16* __restrict__ Qh,
                              _Float16* __restrict__ Kh,
                              _Float16* __restrict__ Vt,
                              float* __restrict__ skip) {
  int idx = blockIdx.x * blockDim.x + threadIdx.x;
  if (idx >= BDIM * NDIM) return;
  int b = idx / NDIM, i = idx % NDIM;
  float x = node[idx];
#pragma unroll 4
  for (int d = 0; d < DDIM; ++d) {
    Qh[(size_t)idx * DDIM + d] = (_Float16)(x * Wq[d]);
    Kh[(size_t)idx * DDIM + d] = (_Float16)(x * Wk[d]);
    Vt[((size_t)b * DDIM + d) * NDIM + i] = (_Float16)(x * Wv[d]);
    skip[(size_t)idx * DDIM + d] = x * Ws[d];
  }
}

// ---------------------------------------------------------------------------
// Projection GEMM: Y[b, n, c] = sum_f X[b, n, f] * W[f, c], fin = 128.
// Wt is the f16-transposed weight (C x 128). One wave = one 16x16 tile.
// MODE 0: f16 row-major; MODE 1: f16 transposed (V); MODE 2: fp32 row-major.
// ---------------------------------------------------------------------------
template <int C, int MODE>
__global__ void __launch_bounds__(32)
proj_gemm_kernel(const float* __restrict__ X, const _Float16* __restrict__ Wt,
                 _Float16* __restrict__ Yh, float* __restrict__ Yf) {
  const int i0 = blockIdx.x * 16;
  const int c0 = blockIdx.y * 16;
  const int b  = blockIdx.z;
  const int lane = threadIdx.x & 31;
  const int nn = lane & 15, hig = lane >> 4;

  const float*    xp = X  + ((size_t)b * NDIM + i0) * DDIM + nn * DDIM + hig * 8;
  const _Float16* wp = Wt + (size_t)c0 * DDIM + nn * DDIM + hig * 8;

  v8f acc = zero_v8f();
#pragma unroll
  for (int ks = 0; ks < 4; ++ks)
    acc = wmma16(ldfrag_pre_f32(xp + ks * 32), ldfrag_pre(wp + ks * 32), acc);

#pragma unroll
  for (int v = 0; v < 8; ++v) {
    float val = acc[v];
    if (MODE == 0) {
      _Float16* yb = Yh + ((size_t)b * NDIM + i0 + 8 * hig) * C + c0 + nn;
      yb[(size_t)v * C] = (_Float16)val;
    } else if (MODE == 1) {
      _Float16* yb = Yh + ((size_t)b * C + c0 + nn) * NDIM + i0 + 8 * hig;
      yb[v] = (_Float16)val;
    } else {
      float* yb = Yf + ((size_t)b * NDIM + i0 + 8 * hig) * C + c0 + nn;
      yb[(size_t)v * C] = val;
    }
  }
}

// ---------------------------------------------------------------------------
// qe[b, h, i] = sum_d Q[b, i, h, d] * We[h, d]
// ---------------------------------------------------------------------------
__global__ void qe_kernel(const _Float16* __restrict__ Qh,
                          const float* __restrict__ We,
                          float* __restrict__ qe, int H) {
  int idx = blockIdx.x * blockDim.x + threadIdx.x;
  if (idx >= BDIM * H * NDIM) return;
  int i = idx % NDIM;
  int h = (idx / NDIM) % H;
  int b = idx / (NDIM * H);
  const int C = H * DDIM;
  const _Float16* q = Qh + ((size_t)b * NDIM + i) * C + (size_t)h * DDIM;
  float s = 0.f;
#pragma unroll 4
  for (int d = 0; d < DDIM; ++d) s += (float)q[d] * We[h * DDIM + d];
  qe[idx] = s;  // idx == (b*H + h)*N + i
}

// ---------------------------------------------------------------------------
// Fused masked flash-attention with edge bias per (b, h, 16-row tile).
// One wave per block; 16 WMMAs per 32-column tile. H is compile-time so all
// strides fold into immediate offsets; running pointers avoid address mults.
// ---------------------------------------------------------------------------
template <int H>
__global__ void __launch_bounds__(32)
attn_kernel(const _Float16* __restrict__ Qh, const _Float16* __restrict__ Kh,
            const _Float16* __restrict__ Vt, const float* __restrict__ e_t,
            const float* __restrict__ qe, const float* __restrict__ We,
            float* __restrict__ O) {
  constexpr int C = H * DDIM;
  const int i0 = blockIdx.x * 16;
  const int h  = blockIdx.y;
  const int b  = blockIdx.z;
  const int lane = threadIdx.x & 31;
  const int nn = lane & 15, hig = lane >> 4;
  const float rs  = 0.088388347648318447f;  // 1/sqrt(128)
  const float NEG = -__builtin_inff();

  __shared__ __align__(16) _Float16 lalpha[16 * 32];

  // Per-lane fragment offsets (loop invariant).
  const int loffC  = nn * C + hig * 8;
  const int loffN  = nn * NDIM + hig * 8;
  const int loff32 = nn * 32 + hig * 8;

  // Q fragments (A operand of QK^T).
  const _Float16* qp = Qh + ((size_t)b * NDIM + i0) * C + h * DDIM + loffC;
  v16h qf[4];
#pragma unroll
  for (int ks = 0; ks < 4; ++ks) qf[ks] = ldfrag_pre(qp + ks * 32);

  const float* qeb = qe + ((size_t)b * H + h) * NDIM + i0 + 8 * hig;
  float qe_r[8];
#pragma unroll
  for (int v = 0; v < 8; ++v) qe_r[v] = qeb[v];

  v8f acc[8];
#pragma unroll
  for (int t = 0; t < 8; ++t) acc[t] = zero_v8f();
  float mrun[8], lrun[8], ewrun[8];
#pragma unroll
  for (int v = 0; v < 8; ++v) { mrun[v] = NEG; lrun[v] = 0.f; ewrun[v] = 0.f; }

  // Running pointers for the j loop.
  const _Float16* kp = Kh + ((size_t)b * NDIM) * C + h * DDIM + loffC;
  const _Float16* vp = Vt + ((size_t)b * C + h * DDIM) * NDIM + loffN;
  const float*    ep = e_t + ((size_t)b * NDIM + i0 + 8 * hig) * NDIM + nn;

#pragma unroll 1
  for (int j0 = 0; j0 < NDIM; j0 += 32) {
    __builtin_prefetch((const void*)(kp + 32 * C), 0, 1);  // next K tile

    // --- scores: two 16-wide sub-tiles, K = 128 contraction ---
    v8f s0 = zero_v8f(), s1 = zero_v8f();
#pragma unroll
    for (int ks = 0; ks < 4; ++ks) {
      s0 = wmma16(qf[ks], ldfrag_pre(kp + ks * 32), s0);
      s1 = wmma16(qf[ks], ldfrag_pre(kp + 16 * C + ks * 32), s1);
    }

    // --- edge bias + mask (immediate-offset e_t loads) ---
    float p0[8], p1[8], et0[8], et1[8], tm[8];
#pragma unroll
    for (int v = 0; v < 8; ++v) {
      float a0 = ep[(size_t)v * NDIM];
      float a1 = ep[(size_t)v * NDIM + 16];
      et0[v] = a0; et1[v] = a1;
      float x0 = (s0[v] + qe_r[v] * a0) * rs;
      float x1 = (s1[v] + qe_r[v] * a1) * rs;
      p0[v] = (a0 != 0.f) ? x0 : NEG;
      p1[v] = (a1 != 0.f) ? x1 : NEG;
      tm[v] = fmaxf(p0[v], p1[v]);
    }
#pragma unroll
    for (int v = 0; v < 8; ++v) {
      float t = tm[v];
      t = fmaxf(t, __shfl_xor(t, 1));
      t = fmaxf(t, __shfl_xor(t, 2));
      t = fmaxf(t, __shfl_xor(t, 4));
      t = fmaxf(t, __shfl_xor(t, 8));
      tm[v] = t;
    }
    // --- online softmax update ---
    float scl[8];
#pragma unroll
    for (int v = 0; v < 8; ++v) {
      float mnew = fmaxf(mrun[v], tm[v]);
      float sc, q0, q1;
      if (mnew == NEG) { sc = 1.f; q0 = 0.f; q1 = 0.f; }
      else {
        sc = __expf(mrun[v] - mnew);  // mrun == -inf -> 0
        q0 = (p0[v] == NEG) ? 0.f : __expf(p0[v] - mnew);
        q1 = (p1[v] == NEG) ? 0.f : __expf(p1[v] - mnew);
      }
      mrun[v] = mnew;
      float rsum = q0 + q1;
      float esum = q0 * et0[v] + q1 * et1[v];
      rsum += __shfl_xor(rsum, 1); rsum += __shfl_xor(rsum, 2);
      rsum += __shfl_xor(rsum, 4); rsum += __shfl_xor(rsum, 8);
      esum += __shfl_xor(esum, 1); esum += __shfl_xor(esum, 2);
      esum += __shfl_xor(esum, 4); esum += __shfl_xor(esum, 8);
      lrun[v]  = lrun[v]  * sc + rsum;
      ewrun[v] = ewrun[v] * sc + esum;
      scl[v] = sc;
      p0[v] = q0; p1[v] = q1;
    }
#pragma unroll
    for (int t = 0; t < 8; ++t)
#pragma unroll
      for (int v = 0; v < 8; ++v) acc[t][v] *= scl[v];

    // --- alpha (C/D layout) -> LDS -> A-operand layout ---
#pragma unroll
    for (int v = 0; v < 8; ++v) {
      int m = v + 8 * hig;
      lalpha[m * 32 + nn]      = (_Float16)p0[v];
      lalpha[m * 32 + 16 + nn] = (_Float16)p1[v];
    }
    __syncthreads();
    v16h af = ldfrag_pre(lalpha + loff32);
    __syncthreads();

    // --- O += alpha(16x32) * V(32x128): 8 WMMAs over d slices ---
#pragma unroll
    for (int t = 0; t < 8; ++t)
      acc[t] = wmma16(af, ldfrag_pre(vp + (size_t)(t * 16) * NDIM), acc[t]);

    kp += 32 * C;
    vp += 32;
    ep += 32;
  }

  // --- finalize: divide by lsum, add edge-weight * We ---
  float we_r[8], inv[8];
#pragma unroll
  for (int t = 0; t < 8; ++t) we_r[t] = We[h * DDIM + t * 16 + nn];
#pragma unroll
  for (int v = 0; v < 8; ++v) inv[v] = (lrun[v] > 0.f) ? (1.f / lrun[v]) : 0.f;
  float* ob = O + (((size_t)b * NDIM + i0 + 8 * hig) * H + h) * DDIM + nn;
#pragma unroll
  for (int t = 0; t < 8; ++t) {
#pragma unroll
    for (int v = 0; v < 8; ++v)
      ob[(size_t)v * H * DDIM + t * 16] =
          (acc[t][v] + ewrun[v] * we_r[t]) * inv[v];
  }
}

// ---------------------------------------------------------------------------
// Epilogue: head mean/concat + skip (+ LayerNorm + ReLU). One wave per row.
// ---------------------------------------------------------------------------
__global__ void __launch_bounds__(32)
combine_kernel(const float* __restrict__ attnO, const float* __restrict__ skip,
               float* __restrict__ out, int H, int do_ln) {
  const int row = blockIdx.x;  // b*N + i
  const int lane = threadIdx.x & 31;
  float v[4];
#pragma unroll
  for (int u = 0; u < 4; ++u) {
    int d = u * 32 + lane;
    float a;
    if (H == 1) a = attnO[(size_t)row * DDIM + d];
    else a = 0.5f * (attnO[((size_t)row * 2 + 0) * DDIM + d] +
                     attnO[((size_t)row * 2 + 1) * DDIM + d]);
    v[u] = a + skip[(size_t)row * DDIM + d];
  }
  if (do_ln) {
    float s = v[0] + v[1] + v[2] + v[3];
    s += __shfl_xor(s, 1); s += __shfl_xor(s, 2); s += __shfl_xor(s, 4);
    s += __shfl_xor(s, 8); s += __shfl_xor(s, 16);
    float mu = s * (1.f / 128.f);
    float q = 0.f;
#pragma unroll
    for (int u = 0; u < 4; ++u) { float d0 = v[u] - mu; q += d0 * d0; }
    q += __shfl_xor(q, 1); q += __shfl_xor(q, 2); q += __shfl_xor(q, 4);
    q += __shfl_xor(q, 8); q += __shfl_xor(q, 16);
    float invs = rsqrtf(q * (1.f / 128.f) + 1e-5f);
#pragma unroll
    for (int u = 0; u < 4; ++u) v[u] = fmaxf(0.f, (v[u] - mu) * invs);
  }
#pragma unroll
  for (int u = 0; u < 4; ++u) out[(size_t)row * DDIM + u * 32 + lane] = v[u];
}

// ---------------------------------------------------------------------------
extern "C" void kernel_launch(void* const* d_in, const int* in_sizes, int n_in,
                              void* d_out, int out_size, void* d_ws,
                              size_t ws_size, hipStream_t stream) {
  (void)in_sizes; (void)n_in; (void)out_size; (void)ws_size;
  const float* node = (const float*)d_in[0];
  const float* edge = (const float*)d_in[1];
  const float* Amat = (const float*)d_in[2];
  const float* Wq[3] = {(const float*)d_in[3],  (const float*)d_in[8],  (const float*)d_in[13]};
  const float* Wk[3] = {(const float*)d_in[4],  (const float*)d_in[9],  (const float*)d_in[14]};
  const float* Wv[3] = {(const float*)d_in[5],  (const float*)d_in[10], (const float*)d_in[15]};
  const float* We[3] = {(const float*)d_in[6],  (const float*)d_in[11], (const float*)d_in[16]};
  const float* Ws[3] = {(const float*)d_in[7],  (const float*)d_in[12], (const float*)d_in[17]};

  char* w = (char*)d_ws;
  auto carve = [&](size_t bytes) {
    void* p = (void*)w;
    w += (bytes + 255) & ~(size_t)255;
    return p;
  };
  float*    e_t  = (float*)   carve((size_t)BDIM * NDIM * NDIM * 4);     // 33.5 MB
  _Float16* Qh   = (_Float16*)carve((size_t)BDIM * NDIM * 256 * 2);      // 2 MB
  _Float16* Kh   = (_Float16*)carve((size_t)BDIM * NDIM * 256 * 2);      // 2 MB
  _Float16* Vt   = (_Float16*)carve((size_t)BDIM * 256 * NDIM * 2);      // 2 MB
  float*    skip = (float*)   carve((size_t)BDIM * NDIM * DDIM * 4);     // 2 MB
  float*    qe   = (float*)   carve((size_t)BDIM * 2 * NDIM * 4);        // 32 KB
  float*    aO   = (float*)   carve((size_t)BDIM * NDIM * 2 * DDIM * 4); // 4 MB
  float*    emb  = (float*)   carve((size_t)BDIM * NDIM * DDIM * 4);     // 2 MB
  // f16-transposed weights (layers 1 and 2)
  _Float16* Wt1q = (_Float16*)carve((size_t)DDIM * DDIM * 2);
  _Float16* Wt1k = (_Float16*)carve((size_t)DDIM * DDIM * 2);
  _Float16* Wt1v = (_Float16*)carve((size_t)DDIM * DDIM * 2);
  _Float16* Wt1s = (_Float16*)carve((size_t)DDIM * DDIM * 2);
  _Float16* Wt2q = (_Float16*)carve((size_t)256 * DDIM * 2);
  _Float16* Wt2k = (_Float16*)carve((size_t)256 * DDIM * 2);
  _Float16* Wt2v = (_Float16*)carve((size_t)256 * DDIM * 2);
  _Float16* Wt2s = (_Float16*)carve((size_t)DDIM * DDIM * 2);
  float* outp = (float*)d_out;

  // e_t = transpose(edge * A)
  {
    size_t tot = (size_t)BDIM * NDIM * NDIM;
    prep_et_kernel<<<dim3((unsigned)((tot + 255) / 256)), 256, 0, stream>>>(edge, Amat, e_t);
  }
  // Weight transposes (once; reused across the 3 recurrences)
  {
    int n1 = DDIM * DDIM, n2 = 256 * DDIM;
    prep_wt_kernel<<<(n1 + 255) / 256, 256, 0, stream>>>(Wq[1], Wt1q, 128);
    prep_wt_kernel<<<(n1 + 255) / 256, 256, 0, stream>>>(Wk[1], Wt1k, 128);
    prep_wt_kernel<<<(n1 + 255) / 256, 256, 0, stream>>>(Wv[1], Wt1v, 128);
    prep_wt_kernel<<<(n1 + 255) / 256, 256, 0, stream>>>(Ws[1], Wt1s, 128);
    prep_wt_kernel<<<(n2 + 255) / 256, 256, 0, stream>>>(Wq[2], Wt2q, 256);
    prep_wt_kernel<<<(n2 + 255) / 256, 256, 0, stream>>>(Wk[2], Wt2k, 256);
    prep_wt_kernel<<<(n2 + 255) / 256, 256, 0, stream>>>(Wv[2], Wt2v, 256);
    prep_wt_kernel<<<(n1 + 255) / 256, 256, 0, stream>>>(Ws[2], Wt2s, 128);
  }

  // ----- layer 0 (fin=1, H=1) -----
  layer0_kernel<<<(BDIM * NDIM + 63) / 64, 64, 0, stream>>>(
      node, Wq[0], Wk[0], Wv[0], Ws[0], Qh, Kh, Vt, skip);
  qe_kernel<<<(BDIM * 1 * NDIM + 63) / 64, 64, 0, stream>>>(Qh, We[0], qe, 1);
  attn_kernel<1><<<dim3(NDIM / 16, 1, BDIM), 32, 0, stream>>>(
      Qh, Kh, Vt, e_t, qe, We[0], aO);
  combine_kernel<<<BDIM * NDIM, 32, 0, stream>>>(aO, skip, emb, 1, 1);

  // ----- recurrences: layer1 (H=1) then layer2 (H=2) -----
  for (int r = 0; r < 3; ++r) {
    // layer 1: C = 128
    dim3 g1(NDIM / 16, DDIM / 16, BDIM);
    proj_gemm_kernel<128, 0><<<g1, 32, 0, stream>>>(emb, Wt1q, Qh, nullptr);
    proj_gemm_kernel<128, 0><<<g1, 32, 0, stream>>>(emb, Wt1k, Kh, nullptr);
    proj_gemm_kernel<128, 1><<<g1, 32, 0, stream>>>(emb, Wt1v, Vt, nullptr);
    proj_gemm_kernel<128, 2><<<g1, 32, 0, stream>>>(emb, Wt1s, nullptr, skip);
    qe_kernel<<<(BDIM * 1 * NDIM + 63) / 64, 64, 0, stream>>>(Qh, We[1], qe, 1);
    attn_kernel<1><<<dim3(NDIM / 16, 1, BDIM), 32, 0, stream>>>(
        Qh, Kh, Vt, e_t, qe, We[1], aO);
    combine_kernel<<<BDIM * NDIM, 32, 0, stream>>>(aO, skip, emb, 1, 1);

    // layer 2: C = 256 (H=2), skip output is 128 wide
    dim3 g2(NDIM / 16, 256 / 16, BDIM);
    proj_gemm_kernel<256, 0><<<g2, 32, 0, stream>>>(emb, Wt2q, Qh, nullptr);
    proj_gemm_kernel<256, 0><<<g2, 32, 0, stream>>>(emb, Wt2k, Kh, nullptr);
    proj_gemm_kernel<256, 1><<<g2, 32, 0, stream>>>(emb, Wt2v, Vt, nullptr);
    proj_gemm_kernel<128, 2><<<g1, 32, 0, stream>>>(emb, Wt2s, nullptr, skip);
    qe_kernel<<<(BDIM * 2 * NDIM + 63) / 64, 64, 0, stream>>>(Qh, We[2], qe, 2);
    attn_kernel<2><<<dim3(NDIM / 16, 2, BDIM), 32, 0, stream>>>(
        Qh, Kh, Vt, e_t, qe, We[2], aO);
    float* dst = (r == 2) ? outp : emb;
    combine_kernel<<<BDIM * NDIM, 32, 0, stream>>>(aO, skip, dst, 2, 0);
  }
}
